// QLSTM_65481071409523
// MI455X (gfx1250) — matrix operand order repaired and checked by reading, not compiled, not run on hardware
//
#include <hip/hip_runtime.h>
#include <hip/hip_bf16.h>

// ---------------------------------------------------------------------------
// QLSTM for MI455X (gfx1250): bf16 WMMA GEMMs (f32 accum), x-part hoisted.
// T=256, B=64, D=H=1024, DEPTH=2.
// Final: rolled (#pragma unroll 1) ping-pong fragment pipeline — steady state
// issues the next k-block's 10-load clause before the current 4 WMMAs, giving
// partial s_wait_loadcnt overlap (best codegen of the variants probed).
// ---------------------------------------------------------------------------

typedef __bf16 bf16;
typedef __attribute__((ext_vector_type(16))) __bf16 v16bf;
typedef __attribute__((ext_vector_type(8)))  float  v8f;

#define T_STEPS 256
#define BATCH   64
#define DDIM    1024
#define HDIM    1024

union ABfrag { uint4 u[2]; v16bf v; };

template <int NT>
struct Frags { ABfrag a; ABfrag b[NT]; };

// Load one A fragment (16x32 bf16) + NT B fragments (32x16 bf16 each).
// Layouts used here (wave32, contiguous k order for both A and B):
//   A: lane<16 holds row M=lane, K {0..15}; lane>=16 K {16..31} (contiguous 32B)
//   B: lane n holds W[n0+n][k0 + (lane>>4)*16 .. +15] (contiguous 32B)
template <int NT>
__device__ __forceinline__ void load_frags(Frags<NT>& f,
                                           const bf16* __restrict__ arow,
                                           const bf16* __restrict__ wrow,
                                           int ldw, int k0) {
    const uint4* ap = reinterpret_cast<const uint4*>(arow + k0);
    f.a.u[0] = ap[0];      // 8 bf16
    f.a.u[1] = ap[1];      // 8 bf16 at +8 elements
#pragma unroll
    for (int j = 0; j < NT; ++j) {
        const uint4* bp = reinterpret_cast<const uint4*>(
            wrow + (size_t)j * 16 * ldw + k0);
        f.b[j].u[0] = bp[0];
        f.b[j].u[1] = bp[1];
    }
}

template <int NT>
__device__ __forceinline__ void wmma_all(const Frags<NT>& f, v8f* acc) {
#pragma unroll
    for (int j = 0; j < NT; ++j)
        acc[j] = __builtin_amdgcn_wmma_f32_16x16x32_bf16(
            false, f.a.v, false, f.b[j].v, (short)0, acc[j], false, false);
}

// One wave computes NT adjacent 16x16 output tiles of  out = A * W^T
//   A: [M,K] row-major bf16 (lda), W: [N,K] row-major bf16 (ldw)
// K must be a multiple of 64. Double-buffered fragment pipeline, kept rolled.
template <int NT>
__device__ __forceinline__ void wave_gemm(const bf16* __restrict__ A, int lda,
                                          const bf16* __restrict__ W, int ldw,
                                          int m0, int n0, int K, v8f* acc) {
    const int lane  = threadIdx.x & 31;
    const int row16 = lane & 15;   // M row (A) / N col (B,C)
    const int khalf = lane >> 4;   // K-half selector
    // fold per-lane offsets into the base pointers once
    const bf16* arow = A + (size_t)(m0 + row16) * lda + khalf * 16;
    const bf16* wrow = W + (size_t)(n0 + row16) * ldw + khalf * 16;

    Frags<NT> f0, f1;
    load_frags<NT>(f0, arow, wrow, ldw, 0);
    int k0 = 0;
#pragma unroll 1
    for (; k0 + 64 < K; k0 += 64) {
        load_frags<NT>(f1, arow, wrow, ldw, k0 + 32);
        wmma_all<NT>(f0, acc);
        load_frags<NT>(f0, arow, wrow, ldw, k0 + 64);
        wmma_all<NT>(f1, acc);
    }
    load_frags<NT>(f1, arow, wrow, ldw, k0 + 32);
    wmma_all<NT>(f0, acc);
    wmma_all<NT>(f1, acc);
}

// ---------------- prep kernels ----------------

__global__ __launch_bounds__(256) void conv_w_kernel(
    const float* __restrict__ Wf, const float* __restrict__ Wi,
    const float* __restrict__ Wu, const float* __restrict__ Wo,
    bf16* __restrict__ Wbf) {           // [4][1024][2048]
    size_t idx = (size_t)blockIdx.x * 256 + threadIdx.x;   // 4 * 2^21
    int g = (int)(idx >> 21);
    size_t off = idx & ((1u << 21) - 1);
    const float* src = (g == 0) ? Wf : (g == 1) ? Wi : (g == 2) ? Wu : Wo;
    Wbf[idx] = (bf16)src[off];
}

__global__ __launch_bounds__(256) void conv_x_kernel(
    const float* __restrict__ x, bf16* __restrict__ xbf) {  // 2^24 elems
    size_t idx = (size_t)blockIdx.x * 256 + threadIdx.x;
    xbf[idx] = (bf16)x[idx];
}

// weight-norm: qcw[row][h] = g[row] * v[row][h] / ||v[row]||,  row = d*1024+o
__global__ __launch_bounds__(256) void qcw_kernel(
    const float* __restrict__ qc_v, const float* __restrict__ qc_g,
    bf16* __restrict__ qcw) {
    int row = blockIdx.x;                    // 0..2047
    const float* v = qc_v + (size_t)row * HDIM;
    __shared__ float red[256];
    float s = 0.f;
    for (int h = threadIdx.x; h < HDIM; h += 256) { float t = v[h]; s += t * t; }
    red[threadIdx.x] = s;
    __syncthreads();
    for (int off = 128; off > 0; off >>= 1) {
        if (threadIdx.x < off) red[threadIdx.x] += red[threadIdx.x + off];
        __syncthreads();
    }
    float scale = qc_g[row] * rsqrtf(red[0]);
    for (int h = threadIdx.x; h < HDIM; h += 256)
        qcw[(size_t)row * HDIM + h] = (bf16)(v[h] * scale);
}

__global__ __launch_bounds__(256) void init_state_kernel(
    bf16* __restrict__ hbf, float* __restrict__ c) {        // 64*1024
    int idx = blockIdx.x * 256 + threadIdx.x;
    hbf[idx] = (bf16)0.f;
    c[idx] = 0.f;
}

// ---------------- stage A: x-part preactivations (parallel over all T) ----
// gx[t*64+b][g*1024+n] = sum_k x[t][b][k]*W_g[n][k] + bias_g[n]
__global__ __launch_bounds__(256) void gemm_x_kernel(
    const bf16* __restrict__ xbf,      // [16384,1024]
    const bf16* __restrict__ Wbf,      // [4][1024][2048] (x cols = 0..1023)
    const float* __restrict__ b0, const float* __restrict__ b1,
    const float* __restrict__ b2, const float* __restrict__ b3,
    float* __restrict__ gx) {          // [16384,4096]
    int wave = (blockIdx.x * 256 + threadIdx.x) >> 5;   // 65536 waves
    int tn4 = wave & 63;        // 64 groups of 4 n-tiles
    int tm  = wave >> 6;        // 1024 m-tiles
    int m0 = tm * 16, n0 = tn4 * 64;
    int gate = n0 >> 10, nloc0 = n0 & 1023;
    const bf16* W = Wbf + (size_t)gate * HDIM * 2048;   // x columns
    v8f acc[4] = {};
    wave_gemm<4>(xbf, DDIM, W, 2048, m0, nloc0, DDIM, acc);
    const float* bias = (gate == 0) ? b0 : (gate == 1) ? b1 : (gate == 2) ? b2 : b3;
    int lane = threadIdx.x & 31;
    int ncol = lane & 15, mh = lane >> 4;
#pragma unroll
    for (int j = 0; j < 4; ++j) {
        float bv = bias[nloc0 + j * 16 + ncol];
        int n = n0 + j * 16 + ncol;
#pragma unroll
        for (int r = 0; r < 8; ++r)
            gx[(size_t)(m0 + r + mh * 8) * 4096 + n] = acc[j][r] + bv;
    }
}

// ---------------- stage B: recurrent h-GEMM + gate activations ------------
// preact = gx_t + h @ Wh^T ; f,i,o=sigmoid, u=tanh ; G0 rows = g*64+b
__global__ __launch_bounds__(256) void gemm_h_kernel(
    const bf16* __restrict__ hbf,      // [64,1024]
    const bf16* __restrict__ Wbf,      // [4][1024][2048] (h cols = 1024..2047)
    const float* __restrict__ gx_t,    // [64,4096]
    bf16* __restrict__ G0) {           // [256,1024]
    int wave = (blockIdx.x * 256 + threadIdx.x) >> 5;   // 256 waves
    int tn4 = wave & 63;
    int tm  = wave >> 6;                                // 0..3
    int m0 = tm * 16, n0 = tn4 * 64;
    int gate = n0 >> 10, nloc0 = n0 & 1023;
    const bf16* W = Wbf + (size_t)gate * HDIM * 2048 + DDIM;  // h columns
    v8f acc[4] = {};
    wave_gemm<4>(hbf, HDIM, W, 2048, m0, nloc0, HDIM, acc);
    int lane = threadIdx.x & 31;
    int ncol = lane & 15, mh = lane >> 4;
#pragma unroll
    for (int j = 0; j < 4; ++j) {
        int nl = nloc0 + j * 16 + ncol;
        int n  = n0 + j * 16 + ncol;
#pragma unroll
        for (int r = 0; r < 8; ++r) {
            int m = m0 + r + mh * 8;       // batch row
            float pre = acc[j][r] + gx_t[(size_t)m * 4096 + n];
            float v = (gate == 2) ? tanhf(pre) : 1.f / (1.f + __expf(-pre));
            G0[(size_t)(gate * 64 + m) * HDIM + nl] = (bf16)v;
        }
    }
}

// ---------------- stage C: qc layer (shared across gates) -----------------
// Gout = tanh(Gin @ qcw_d^T + qcb_d), Gin/Gout: [256,1024]
__global__ __launch_bounds__(256) void qc_kernel(
    const bf16* __restrict__ Gin, const bf16* __restrict__ qcw_d,
    const float* __restrict__ qcb_d, bf16* __restrict__ Gout) {
    int wave = (blockIdx.x * 256 + threadIdx.x) >> 5;   // 256 waves
    int tn4 = wave & 15;        // 16 groups of 4 n-tiles
    int tm  = wave >> 4;        // 16 m-tiles
    int m0 = tm * 16, n0 = tn4 * 64;
    v8f acc[4] = {};
    wave_gemm<4>(Gin, HDIM, qcw_d, HDIM, m0, n0, HDIM, acc);
    int lane = threadIdx.x & 31;
    int ncol = lane & 15, mh = lane >> 4;
#pragma unroll
    for (int j = 0; j < 4; ++j) {
        int n = n0 + j * 16 + ncol;
        float bv = qcb_d[n];
#pragma unroll
        for (int r = 0; r < 8; ++r)
            Gout[(size_t)(m0 + r + mh * 8) * HDIM + n] = (bf16)tanhf(acc[j][r] + bv);
    }
}

// ---------------- stage D: cell/hidden update -----------------------------
__global__ __launch_bounds__(256) void cell_kernel(
    const bf16* __restrict__ G,        // [256,1024] = f,i,u,o stacked
    float* __restrict__ c, bf16* __restrict__ hbf,
    float* __restrict__ out_t, float* __restrict__ hx, float* __restrict__ cx,
    int last) {
    int idx = blockIdx.x * 256 + threadIdx.x;           // 65536
    float f = (float)G[idx];
    float i = (float)G[65536 + idx];
    float u = (float)G[131072 + idx];
    float o = (float)G[196608 + idx];
    float cn = f * c[idx] + i * u;
    float h = o * tanhf(cn);
    c[idx] = cn;
    hbf[idx] = (bf16)h;
    out_t[idx] = h;
    if (last) { hx[idx] = h; cx[idx] = cn; }
}

// ---------------------------------------------------------------------------

extern "C" void kernel_launch(void* const* d_in, const int* in_sizes, int n_in,
                              void* d_out, int out_size, void* d_ws, size_t ws_size,
                              hipStream_t stream) {
    const float* inputs = (const float*)d_in[0];
    const float* Wf = (const float*)d_in[1];  const float* bf_ = (const float*)d_in[2];
    const float* Wi = (const float*)d_in[3];  const float* bi_ = (const float*)d_in[4];
    const float* Wu = (const float*)d_in[5];  const float* bu_ = (const float*)d_in[6];
    const float* Wo = (const float*)d_in[7];  const float* bo_ = (const float*)d_in[8];
    const float* qcv = (const float*)d_in[9];
    const float* qcg = (const float*)d_in[10];
    const float* qcb = (const float*)d_in[11];
    float* out = (float*)d_out;

    char* ws = (char*)d_ws;
    bf16*  Wbf = (bf16*)(ws);                              // 16 MB  [4][1024][2048]
    bf16*  qcw = (bf16*)(ws + (16ull << 20));              //  4 MB  [2][1024][1024]
    bf16*  xbf = (bf16*)(ws + (20ull << 20));              // 32 MB  [16384][1024]
    float* gx  = (float*)(ws + (52ull << 20));             // 256 MB [16384][4096]
    char* tail = ws + (308ull << 20);
    bf16*  G0  = (bf16*)tail;               tail += 256 * 1024 * sizeof(bf16);
    bf16*  G1  = (bf16*)tail;               tail += 256 * 1024 * sizeof(bf16);
    bf16*  hbf = (bf16*)tail;               tail += 64 * 1024 * sizeof(bf16);
    float* cst = (float*)tail;

    // one-time (per call) prep: bf16 weights, weight-norm qc weights, bf16 x
    conv_w_kernel<<<32768, 256, 0, stream>>>(Wf, Wi, Wu, Wo, Wbf);
    qcw_kernel<<<2048, 256, 0, stream>>>(qcv, qcg, qcw);
    conv_x_kernel<<<65536, 256, 0, stream>>>(inputs, xbf);
    init_state_kernel<<<256, 256, 0, stream>>>(hbf, cst);

    // hoisted x-part GEMM for all timesteps (parallel, WMMA)
    gemm_x_kernel<<<8192, 256, 0, stream>>>(xbf, Wbf, bf_, bi_, bu_, bo_, gx);

    float* hx = out + (size_t)T_STEPS * BATCH * HDIM;
    float* cx = hx + (size_t)BATCH * HDIM;
    for (int t = 0; t < T_STEPS; ++t) {
        gemm_h_kernel<<<32, 256, 0, stream>>>(hbf, Wbf, gx + (size_t)t * BATCH * 4096, G0);
        qc_kernel<<<32, 256, 0, stream>>>(G0, qcw, qcb, G1);
        qc_kernel<<<32, 256, 0, stream>>>(G1, qcw + 1024 * 1024, qcb + 1024, G0);
        cell_kernel<<<256, 256, 0, stream>>>(G0, cst, hbf,
                                             out + (size_t)t * BATCH * HDIM,
                                             hx, cx, t == T_STEPS - 1);
    }
}
